// Model_57397942943850
// MI455X (gfx1250) — compile-verified, matrix operand order checked by reading, not run on hardware
//
#include <hip/hip_runtime.h>
#include <hip/hip_bf16.h>
#include <math.h>

// ---------------- problem constants ----------------
#define BB    8
#define SEQ   512
#define DEC   512
#define DMODEL 512
#define NH    8
#define DH    64
#define DFF   2048
#define CIN   21
#define ELAY  2
#define UU    35
#define NTOP  35
#define PRED  256
#define NEGV  (-1.0e9f)

typedef __attribute__((ext_vector_type(16))) _Float16 v16h;
typedef __attribute__((ext_vector_type(8)))  float    v8f;

// =====================================================================
// WMMA GEMM:  C[M,N] = act( A[M,K] @ W[K,N] + bias )
// fp32 in/out, f16 WMMA with f32 accumulation.
// Block: 256 threads (8 waves). Tile BM=64 x BN=128 x BK=32.
// Wave -> m-tile = wave&3, n-group = wave>>2 (4 n-tiles of 16).
// Each wave: 1 A fragment reused across 4 WMMAs per K-chunk.
// =====================================================================
#define BM 64
#define BN 128
#define BK 32
#define LDSS 34   // f16 stride with padding (17 dwords, odd -> conflict free)

__global__ __launch_bounds__(256) void wmma_gemm_kernel(
    const float* __restrict__ A, const float* __restrict__ W,
    const float* __restrict__ bias, float* __restrict__ C,
    int M, int N, int K, int act)
{
    __shared__ _Float16 As[BM * LDSS];
    __shared__ _Float16 Bs[BN * LDSS];

    const int tid  = threadIdx.x;
    const int wave = tid >> 5;
    const int lane = tid & 31;
    const int g    = lane >> 4;   // half-wave group
    const int mr   = lane & 15;

    const int m0 = blockIdx.y * BM;
    const int n0 = blockIdx.x * BN;

    const int tm  = wave & 3;          // m-tile for this wave (0..3)
    const int tn0 = (wave >> 2) << 2;  // first of four n-tiles (0 or 4)

    v8f acc[4] = {v8f{}, v8f{}, v8f{}, v8f{}};

    for (int k0 = 0; k0 < K; k0 += BK) {
        // ---- stage A tile (64x32) ----
#pragma unroll
        for (int i = 0; i < 8; ++i) {
            int lin = tid + i * 256;
            int r = lin >> 5, c = lin & 31;
            As[r * LDSS + c] = (_Float16)A[(size_t)(m0 + r) * K + (k0 + c)];
        }
        // ---- stage B tile (32x128), transposed to [n][k] ----
#pragma unroll
        for (int i = 0; i < 16; ++i) {
            int lin = tid + i * 256;
            int kk = lin >> 7;       // 0..31
            int nn = lin & 127;      // 0..127
            float v = 0.0f;
            if (n0 + nn < N) v = W[(size_t)(k0 + kk) * N + (n0 + nn)];
            Bs[nn * LDSS + kk] = (_Float16)v;
        }
        // ---- prefetch next K-chunk into cache (global_prefetch_b8) ----
        if (k0 + BK < K) {
            {
                int lin = tid;
                int r = lin >> 5, c = lin & 31;
                __builtin_prefetch(&A[(size_t)(m0 + r) * K + (k0 + BK + c)], 0, 1);
            }
            {
                int kk = tid >> 3;          // 0..31
                int nn = (tid & 7) << 4;    // 0,16,...,112
                if (n0 + nn < N)
                    __builtin_prefetch(&W[(size_t)(k0 + BK + kk) * N + (n0 + nn)], 0, 1);
            }
        }
        __syncthreads();

        // ---- build fragments per ISA 7.12.2 layouts ----
        v16h af, bf[4];
#pragma unroll
        for (int j = 0; j < 8; ++j) {
            int kk = ((j >> 2) << 4) + (g << 3) + ((j & 3) << 1);
            af[2 * j]     = As[(tm * 16 + mr) * LDSS + kk];
            af[2 * j + 1] = As[(tm * 16 + mr) * LDSS + kk + 1];
#pragma unroll
            for (int tt = 0; tt < 4; ++tt) {
                bf[tt][2 * j]     = Bs[((tn0 + tt) * 16 + mr) * LDSS + kk];
                bf[tt][2 * j + 1] = Bs[((tn0 + tt) * 16 + mr) * LDSS + kk + 1];
            }
        }

#pragma unroll
        for (int tt = 0; tt < 4; ++tt)
            acc[tt] = __builtin_amdgcn_wmma_f32_16x16x32_f16(
                false, af, false, bf[tt], (short)0, acc[tt], false, false);
        __syncthreads();
    }

    // ---- epilogue: bias + optional exact GELU ----
#pragma unroll
    for (int tt = 0; tt < 4; ++tt) {
        const int n = n0 + (tn0 + tt) * 16 + mr;
        if (n < N) {
            float bv = bias ? bias[n] : 0.0f;
#pragma unroll
            for (int r = 0; r < 8; ++r) {
                int m = m0 + tm * 16 + r + (g << 3);
                float v = acc[tt][r] + bv;
                if (act == 1) v = 0.5f * v * (1.0f + erff(v * 0.70710678118654752f));
                C[(size_t)m * N + n] = v;
            }
        }
    }
}

// =====================================================================
// Embedding: circular conv-3 token embed + sinusoidal PE + time-mark proj
// one block per (b,l) row, 256 threads over DMODEL
// =====================================================================
__global__ __launch_bounds__(256) void embed_kernel(
    const float* __restrict__ x, const float* __restrict__ mark,
    const float* __restrict__ tokw, const float* __restrict__ timew,
    float* __restrict__ out, int L)
{
    const int row = blockIdx.x;          // b*L + l
    const int b = row / L, l = row % L;
    const int t = threadIdx.x;

    __shared__ float xl[64];
    __shared__ float mk[4];
    if (t < 63) {
        int k = t / CIN, c = t % CIN;
        int src = l + k - 1;
        if (src < 0) src += L;
        if (src >= L) src -= L;
        xl[t] = x[(size_t)(b * L + src) * CIN + c];
    }
    if (t >= 64 && t < 68) mk[t - 64] = mark[(size_t)row * 4 + (t - 64)];
    __syncthreads();

    for (int d = t; d < DMODEL; d += 256) {
        float de  = (float)(d & ~1);
        float dv  = expf(de * (-9.210340371976184f / (float)DMODEL));
        float ang = (float)l * dv;
        float acc = (d & 1) ? cosf(ang) : sinf(ang);
#pragma unroll
        for (int j = 0; j < 4; ++j) acc += mk[j] * timew[j * DMODEL + d];
        for (int kc = 0; kc < 63; ++kc) acc += xl[kc] * tokw[(size_t)kc * DMODEL + d];
        out[(size_t)row * DMODEL + d] = acc;
    }
}

// =====================================================================
// Fused residual + LayerNorm over 512 cols. One block (256 thr) per row.
// =====================================================================
__global__ __launch_bounds__(256) void ln_kernel(
    const float* __restrict__ x, const float* __restrict__ res,
    const float* __restrict__ g, const float* __restrict__ bb,
    float* __restrict__ out)
{
    const int row = blockIdx.x;
    const int t = threadIdx.x;
    __shared__ float red[256];
    const size_t base = (size_t)row * DMODEL;

    float v0 = x[base + t];
    float v1 = x[base + t + 256];
    if (res) { v0 += res[base + t]; v1 += res[base + t + 256]; }

    red[t] = v0 + v1;
    __syncthreads();
    for (int off = 128; off; off >>= 1) {
        if (t < off) red[t] += red[t + off];
        __syncthreads();
    }
    float mean = red[0] * (1.0f / DMODEL);
    __syncthreads();

    float d0 = v0 - mean, d1 = v1 - mean;
    red[t] = d0 * d0 + d1 * d1;
    __syncthreads();
    for (int off = 128; off; off >>= 1) {
        if (t < off) red[t] += red[t + off];
        __syncthreads();
    }
    float inv = rsqrtf(red[0] * (1.0f / DMODEL) + 1e-5f);

    out[base + t]       = d0 * inv * g[t] + bb[t];
    out[base + t + 256] = d1 * inv * g[t + 256] + bb[t + 256];
}

// =====================================================================
// Sampled QK measurement:  M[b,h,l] = max_u(QK_u) - sum_u(QK_u)/LK
// one wave (32 lanes) per (l,h,b); Q,K stored (B,L,H,D)
// =====================================================================
__global__ __launch_bounds__(32) void qk_samp_kernel(
    const float* __restrict__ Q, const float* __restrict__ K,
    const int* __restrict__ samp, float* __restrict__ Mout, int LK)
{
    const int l = blockIdx.x, h = blockIdx.y, b = blockIdx.z;
    const int lane = threadIdx.x;
    const float* qrow = Q + ((size_t)(b * 512 + l) * NH + h) * DH;
    const float q0 = qrow[lane], q1 = qrow[lane + 32];

    float mx = -3.0e38f, sm = 0.0f;
    for (int u = 0; u < UU; ++u) {
        int j = samp[l * UU + u];
        const float* kr = K + ((size_t)(b * LK + j) * NH + h) * DH;
        float p = q0 * kr[lane] + q1 * kr[lane + 32];
#pragma unroll
        for (int off = 16; off; off >>= 1) p += __shfl_xor(p, off, 32);
        mx = fmaxf(mx, p);
        sm += p;
    }
    if (lane == 0)
        Mout[((size_t)b * NH + h) * 512 + l] = mx - sm / (float)LK;
}

// =====================================================================
// top-k (k=35) over 512 with lowest-index tie-break. One block per (b,h).
// =====================================================================
__global__ __launch_bounds__(512) void topk_kernel(
    const float* __restrict__ M, int* __restrict__ idx)
{
    __shared__ float sv[512];
    __shared__ float rv[512];
    __shared__ int   ri[512];
    const int t = threadIdx.x;
    const int bh = blockIdx.x;
    sv[t] = M[(size_t)bh * 512 + t];
    __syncthreads();

    for (int s = 0; s < NTOP; ++s) {
        rv[t] = sv[t];
        ri[t] = t;
        __syncthreads();
        for (int off = 256; off; off >>= 1) {
            if (t < off) {
                float vo = rv[t + off]; int io = ri[t + off];
                if (vo > rv[t] || (vo == rv[t] && io < ri[t])) { rv[t] = vo; ri[t] = io; }
            }
            __syncthreads();
        }
        if (t == 0) { idx[bh * NTOP + s] = ri[0]; sv[ri[0]] = -3.0e38f; }
        __syncthreads();
    }
}

// ===================== ctx initialization ============================
__global__ __launch_bounds__(256) void vmean_kernel(const float* __restrict__ V,
                                                    float* __restrict__ vm, int LK)
{
    int t = blockIdx.x * blockDim.x + threadIdx.x;   // BB*DMODEL threads
    if (t >= BB * DMODEL) return;
    int b = t / DMODEL, hd = t % DMODEL;
    float acc = 0.0f;
    for (int l = 0; l < LK; ++l) acc += V[((size_t)(b * LK + l)) * DMODEL + hd];
    vm[t] = acc / (float)LK;
}

__global__ __launch_bounds__(256) void ctx_fill_kernel(const float* __restrict__ vm,
                                                       float* __restrict__ ctx)
{
    size_t i = (size_t)blockIdx.x * blockDim.x + threadIdx.x;
    if (i >= (size_t)BB * 512 * DMODEL) return;
    int b = (int)(i / ((size_t)512 * DMODEL));
    int hd = (int)(i % DMODEL);
    ctx[i] = vm[b * DMODEL + hd];
}

__global__ __launch_bounds__(256) void cumsum_kernel(const float* __restrict__ V,
                                                     float* __restrict__ ctx)
{
    int t = blockIdx.x * blockDim.x + threadIdx.x;   // BB*DMODEL threads
    if (t >= BB * DMODEL) return;
    int b = t / DMODEL, hd = t % DMODEL;
    float acc = 0.0f;
    for (int l = 0; l < 512; ++l) {
        size_t i = ((size_t)(b * 512 + l)) * DMODEL + hd;
        acc += V[i];
        ctx[i] = acc;
    }
}

// =====================================================================
// Attention update for one selected query: scores -> (mask) -> softmax -> @V
// One block (256 thr) per (u,h,b). Writes ctx row (b,l,h,:).
// =====================================================================
__global__ __launch_bounds__(256) void attn_update_kernel(
    const float* __restrict__ Q, const float* __restrict__ K,
    const float* __restrict__ V, const int* __restrict__ idx,
    float* __restrict__ ctx, int LK, int causal)
{
    const int u = blockIdx.x, h = blockIdx.y, b = blockIdx.z;
    const int t = threadIdx.x;
    __shared__ float qr[DH];
    __shared__ float sc[512];
    __shared__ float red[256];

    const int l = idx[(b * NH + h) * NTOP + u];
    if (t < DH) qr[t] = Q[((size_t)(b * 512 + l) * NH + h) * DH + t];
    __syncthreads();

    float s[2];
#pragma unroll
    for (int half = 0; half < 2; ++half) {
        int kk = t + half * 256;
        const float* kr = K + ((size_t)(b * LK + kk) * NH + h) * DH;
        float acc = 0.0f;
        for (int d = 0; d < DH; ++d) acc += qr[d] * kr[d];
        acc *= 0.125f;                      // 1/sqrt(64)
        if (causal && kk > l) acc = NEGV;
        s[half] = acc;
        sc[kk] = acc;
    }

    // softmax over 512
    red[t] = fmaxf(s[0], s[1]);
    __syncthreads();
    for (int off = 128; off; off >>= 1) {
        if (t < off) red[t] = fmaxf(red[t], red[t + off]);
        __syncthreads();
    }
    float mx = red[0];
    __syncthreads();
    float p0 = expf(s[0] - mx), p1 = expf(s[1] - mx);
    red[t] = p0 + p1;
    __syncthreads();
    for (int off = 128; off; off >>= 1) {
        if (t < off) red[t] += red[t + off];
        __syncthreads();
    }
    float inv = 1.0f / red[0];
    __syncthreads();
    sc[t] = p0 * inv;
    sc[t + 256] = p1 * inv;
    __syncthreads();

    // upd[d] = sum_k p[k] * V[b,k,h,d]
    const int d = t & 63, seg = t >> 6;
    float acc = 0.0f;
    for (int k = seg * 128; k < seg * 128 + 128; ++k)
        acc += sc[k] * V[((size_t)(b * LK + k) * NH + h) * DH + d];
    red[t] = acc;
    __syncthreads();
    if (t < 64) {
        float v = red[t] + red[t + 64] + red[t + 128] + red[t + 192];
        ctx[((size_t)(b * 512 + l) * NH + h) * DH + t] = v;
    }
}

// ===================== final slice =====================
__global__ __launch_bounds__(256) void slice_kernel(const float* __restrict__ full,
                                                    float* __restrict__ out)
{
    int i = blockIdx.x * blockDim.x + threadIdx.x;
    if (i >= BB * PRED * CIN) return;
    int c = i % CIN;
    int p = (i / CIN) % PRED;
    int b = i / (CIN * PRED);
    out[i] = full[((size_t)(b * DEC + (DEC - PRED + p))) * CIN + c];
}

// =====================================================================
// Host-side orchestration
// =====================================================================
static void gemm(const float* A, const float* W, const float* bias, float* C,
                 int M, int N, int K, int act, hipStream_t s)
{
    dim3 grid((N + BN - 1) / BN, (M + BM - 1) / BM);
    wmma_gemm_kernel<<<grid, 256, 0, s>>>(A, W, bias, C, M, N, K, act);
}

struct Ws {
    float *X, *Q, *K, *V, *CTX, *TMP, *ENC, *FF, *POUT, *Mbuf, *VMEAN;
    int* IDX;
};

static void attention_block(const float* xq, const float* xkv,
                            const float* qkv_w, const float* qkv_b,
                            const float* ow, const float* ob,
                            const int* samp, int causal,
                            const Ws& w, hipStream_t s)
{
    const int M = BB * 512;
    gemm(xq,  qkv_w + 0 * DMODEL * DMODEL, qkv_b + 0 * DMODEL, w.Q, M, DMODEL, DMODEL, 0, s);
    gemm(xkv, qkv_w + 1 * DMODEL * DMODEL, qkv_b + 1 * DMODEL, w.K, M, DMODEL, DMODEL, 0, s);
    gemm(xkv, qkv_w + 2 * DMODEL * DMODEL, qkv_b + 2 * DMODEL, w.V, M, DMODEL, DMODEL, 0, s);

    qk_samp_kernel<<<dim3(512, NH, BB), 32, 0, s>>>(w.Q, w.K, samp, w.Mbuf, 512);
    topk_kernel<<<BB * NH, 512, 0, s>>>(w.Mbuf, w.IDX);

    if (causal) {
        cumsum_kernel<<<(BB * DMODEL + 255) / 256, 256, 0, s>>>(w.V, w.CTX);
    } else {
        vmean_kernel<<<(BB * DMODEL + 255) / 256, 256, 0, s>>>(w.V, w.VMEAN, 512);
        ctx_fill_kernel<<<(BB * 512 * DMODEL) / 256, 256, 0, s>>>(w.VMEAN, w.CTX);
    }
    attn_update_kernel<<<dim3(NTOP, NH, BB), 256, 0, s>>>(w.Q, w.K, w.V, w.IDX,
                                                          w.CTX, 512, causal);
    gemm(w.CTX, ow, ob, w.TMP, M, DMODEL, DMODEL, 0, s);
}

extern "C" void kernel_launch(void* const* d_in, const int* in_sizes, int n_in,
                              void* d_out, int out_size, void* d_ws, size_t ws_size,
                              hipStream_t stream)
{
    // -------- inputs (setup_inputs order) --------
    const float* x_enc        = (const float*)d_in[0];
    const float* x_mark_enc   = (const float*)d_in[1];
    const float* x_dec        = (const float*)d_in[2];
    const float* x_mark_dec   = (const float*)d_in[3];
    const float* enc_tok_w    = (const float*)d_in[4];
    const float* dec_tok_w    = (const float*)d_in[5];
    const float* enc_time_w   = (const float*)d_in[6];
    const float* dec_time_w   = (const float*)d_in[7];
    const float* enc_qkv_w    = (const float*)d_in[8];
    const float* enc_qkv_b    = (const float*)d_in[9];
    const float* enc_out_w    = (const float*)d_in[10];
    const float* enc_out_b    = (const float*)d_in[11];
    const float* enc_ff1_w    = (const float*)d_in[12];
    const float* enc_ff1_b    = (const float*)d_in[13];
    const float* enc_ff2_w    = (const float*)d_in[14];
    const float* enc_ff2_b    = (const float*)d_in[15];
    const float* enc_ln_g     = (const float*)d_in[16];
    const float* enc_ln_b     = (const float*)d_in[17];
    const float* enc_norm_g   = (const float*)d_in[18];
    const float* enc_norm_b   = (const float*)d_in[19];
    const float* dec_self_qkv_w = (const float*)d_in[20];
    const float* dec_self_qkv_b = (const float*)d_in[21];
    const float* dec_self_out_w = (const float*)d_in[22];
    const float* dec_self_out_b = (const float*)d_in[23];
    const float* dec_cross_qkv_w = (const float*)d_in[24];
    const float* dec_cross_qkv_b = (const float*)d_in[25];
    const float* dec_cross_out_w = (const float*)d_in[26];
    const float* dec_cross_out_b = (const float*)d_in[27];
    const float* dec_ff1_w    = (const float*)d_in[28];
    const float* dec_ff1_b    = (const float*)d_in[29];
    const float* dec_ff2_w    = (const float*)d_in[30];
    const float* dec_ff2_b    = (const float*)d_in[31];
    const float* dec_ln_g     = (const float*)d_in[32];
    const float* dec_ln_b     = (const float*)d_in[33];
    const float* dec_norm_g   = (const float*)d_in[34];
    const float* dec_norm_b   = (const float*)d_in[35];
    const float* proj_w       = (const float*)d_in[36];
    const float* proj_b       = (const float*)d_in[37];
    const int*   samp_enc     = (const int*)d_in[38];
    const int*   samp_dec_self= (const int*)d_in[39];
    const int*   samp_cross   = (const int*)d_in[40];

    (void)in_sizes; (void)n_in; (void)out_size; (void)ws_size;

    // -------- workspace layout --------
    const size_t ACT = (size_t)BB * 512 * DMODEL;        // 2,097,152 floats
    float* ws = (float*)d_ws;
    Ws w;
    w.X     = ws;
    w.Q     = w.X   + ACT;
    w.K     = w.Q   + ACT;
    w.V     = w.K   + ACT;
    w.CTX   = w.V   + ACT;
    w.TMP   = w.CTX + ACT;
    w.ENC   = w.TMP + ACT;
    w.FF    = w.ENC + ACT;                               // BB*512*DFF
    w.POUT  = w.FF  + (size_t)BB * 512 * DFF;            // BB*512*CIN
    w.Mbuf  = w.POUT + (size_t)BB * 512 * CIN;           // BB*NH*512
    w.VMEAN = w.Mbuf + (size_t)BB * NH * 512;            // BB*DMODEL
    w.IDX   = (int*)(w.VMEAN + (size_t)BB * DMODEL);     // BB*NH*NTOP ints

    const int M = BB * 512;

    // ================= Encoder =================
    embed_kernel<<<BB * SEQ, 256, 0, stream>>>(x_enc, x_mark_enc, enc_tok_w,
                                               enc_time_w, w.X, SEQ);
    for (int l = 0; l < ELAY; ++l) {
        const float* qkv_w = enc_qkv_w + (size_t)l * 3 * DMODEL * DMODEL;
        const float* qkv_b = enc_qkv_b + (size_t)l * 3 * DMODEL;
        const float* ow    = enc_out_w + (size_t)l * DMODEL * DMODEL;
        const float* ob    = enc_out_b + (size_t)l * DMODEL;
        const int*   smp   = samp_enc + (size_t)l * SEQ * UU;

        attention_block(w.X, w.X, qkv_w, qkv_b, ow, ob, smp, 0, w, stream);
        ln_kernel<<<M, 256, 0, stream>>>(w.X, w.TMP,
                                         enc_ln_g + ((size_t)l * 2 + 0) * DMODEL,
                                         enc_ln_b + ((size_t)l * 2 + 0) * DMODEL, w.X);

        gemm(w.X, enc_ff1_w + (size_t)l * DMODEL * DFF, enc_ff1_b + (size_t)l * DFF,
             w.FF, M, DFF, DMODEL, 1, stream);
        gemm(w.FF, enc_ff2_w + (size_t)l * DFF * DMODEL, enc_ff2_b + (size_t)l * DMODEL,
             w.TMP, M, DMODEL, DFF, 0, stream);
        ln_kernel<<<M, 256, 0, stream>>>(w.X, w.TMP,
                                         enc_ln_g + ((size_t)l * 2 + 1) * DMODEL,
                                         enc_ln_b + ((size_t)l * 2 + 1) * DMODEL, w.X);
    }
    ln_kernel<<<M, 256, 0, stream>>>(w.X, nullptr, enc_norm_g, enc_norm_b, w.ENC);

    // ================= Decoder =================
    embed_kernel<<<BB * DEC, 256, 0, stream>>>(x_dec, x_mark_dec, dec_tok_w,
                                               dec_time_w, w.X, DEC);

    // self attention (causal prob-sparse)
    attention_block(w.X, w.X, dec_self_qkv_w, dec_self_qkv_b,
                    dec_self_out_w, dec_self_out_b, samp_dec_self, 1, w, stream);
    ln_kernel<<<M, 256, 0, stream>>>(w.X, w.TMP, dec_ln_g + 0 * DMODEL,
                                     dec_ln_b + 0 * DMODEL, w.X);

    // cross attention (kv from encoder output)
    attention_block(w.X, w.ENC, dec_cross_qkv_w, dec_cross_qkv_b,
                    dec_cross_out_w, dec_cross_out_b, samp_cross, 0, w, stream);
    ln_kernel<<<M, 256, 0, stream>>>(w.X, w.TMP, dec_ln_g + 1 * DMODEL,
                                     dec_ln_b + 1 * DMODEL, w.X);

    // FFN
    gemm(w.X, dec_ff1_w, dec_ff1_b, w.FF, M, DFF, DMODEL, 1, stream);
    gemm(w.FF, dec_ff2_w, dec_ff2_b, w.TMP, M, DMODEL, DFF, 0, stream);
    ln_kernel<<<M, 256, 0, stream>>>(w.X, w.TMP, dec_ln_g + 2 * DMODEL,
                                     dec_ln_b + 2 * DMODEL, w.X);

    // final norm + projection + slice
    ln_kernel<<<M, 256, 0, stream>>>(w.X, nullptr, dec_norm_g, dec_norm_b, w.X);
    gemm(w.X, proj_w, proj_b, w.POUT, M, CIN, DMODEL, 0, stream);
    slice_kernel<<<(BB * PRED * CIN + 255) / 256, 256, 0, stream>>>(w.POUT,
                                                                    (float*)d_out);
}
